// MLP_22565758173774
// MI455X (gfx1250) — compile-verified, matrix operand order checked by reading, not run on hardware
//
#include <hip/hip_runtime.h>
#include <hip/hip_bf16.h>

typedef __attribute__((ext_vector_type(16))) _Float16 v16h;
typedef __attribute__((ext_vector_type(8)))  _Float16 v8h;
typedef __attribute__((ext_vector_type(4)))  _Float16 v4h;
typedef __attribute__((ext_vector_type(8)))  float    v8f;

static constexpr int T_TOK = 8192;
static constexpr int HD    = 4096;
static constexpr int MDIM  = 11008;
static constexpr int QBLK  = 64;    // NF4 quant block
static constexpr int LDK   = 40;    // LDS row stride (32 halves + 8 pad -> conflict-free)
static constexpr int BM    = 128;   // block tile (tokens)
static constexpr int BN    = 128;   // block tile (features)
static constexpr int BK    = 32;

__device__ __constant__ float NF4_TAB[16] = {
    -1.0f, -0.6961928009986877f, -0.5250730514526367f, -0.39491748809814453f,
    -0.28444138169288635f, -0.18477343022823334f, -0.09105003625154495f, 0.0f,
    0.07958029955625534f, 0.16093020141124725f, 0.24611230194568634f,
    0.33791524171829224f, 0.44070982933044434f, 0.5626170039176941f,
    0.7229568362236023f, 1.0f};

// ---------------------------------------------------------------------------
// NF4 dequant: codes int32 [N,K] + scales f32 [N,K/64] -> f16 [N,K]
// ---------------------------------------------------------------------------
__global__ __launch_bounds__(256) void dequant_nf4_kernel(
    const int* __restrict__ codes, const float* __restrict__ scales,
    _Float16* __restrict__ out, int K, int KB)
{
    __shared__ float tab[16];
    if (threadIdx.x < 16) tab[threadIdx.x] = NF4_TAB[threadIdx.x];
    __syncthreads();

    int base = (blockIdx.x * 256 + threadIdx.x) * 4;   // < 45M, fits int
    int4 c = *(const int4*)(codes + base);
    int row = base / K;
    int k   = base - row * K;
    float s = scales[row * KB + (k >> 6)];
    v4h o;
    o[0] = (_Float16)(tab[c.x & 15] * s);
    o[1] = (_Float16)(tab[c.y & 15] * s);
    o[2] = (_Float16)(tab[c.z & 15] * s);
    o[3] = (_Float16)(tab[c.w & 15] * s);
    *(v4h*)(out + base) = o;
}

// ---------------------------------------------------------------------------
// Fragment load matching CDNA5 16-bit A/B WMMA layout:
// lane<16:  row=lane,    K = 0..7  and 16..23
// lane>=16: row=lane-16, K = 8..15 and 24..31
// -> two ds_load_b128, conflict-free with LDK=40 (80B) row stride.
// ---------------------------------------------------------------------------
__device__ __forceinline__ v16h load_frag(const _Float16* tile, int row0, int lane)
{
    int r  = row0 + (lane & 15);
    int kh = (lane >> 4) << 3;               // 0 or 8
    v8h lo = *(const v8h*)(tile + r * LDK + kh);
    v8h hi = *(const v8h*)(tile + r * LDK + 16 + kh);
    v16h out;
#pragma unroll
    for (int i = 0; i < 8; ++i) { out[i] = lo[i]; out[8 + i] = hi[i]; }
    return out;
}

__device__ __forceinline__ v8f wmma16(v16h a, v16h b, v8f c)
{
    return __builtin_amdgcn_wmma_f32_16x16x32_f16(
        false, a, false, b, (short)0, c, false, false);
}

// ---------------------------------------------------------------------------
// Fused gate/up GEMM + SwiGLU.  x f32 [T,HD], Wg/Wu f16 [M,HD] -> h f16 [T,M]
// 256 threads = 8 waves (2x4); block tile 128x128; wave tile 64x32; BK=32.
// Double-buffered LDS, one barrier per K step.
// ---------------------------------------------------------------------------
__global__ __launch_bounds__(256) void gemm_gateup_kernel(
    const float* __restrict__ x, const _Float16* __restrict__ Wg,
    const _Float16* __restrict__ Wu, _Float16* __restrict__ h,
    int T, int K, int M)
{
    __shared__ _Float16 xs[2][BM * LDK];
    __shared__ _Float16 gs[2][BN * LDK];
    __shared__ _Float16 us[2][BN * LDK];

    const int tid  = threadIdx.x;
    const int lane = tid & 31;
    const int wave = tid >> 5;
    const int wm   = (wave >> 2) * 64;       // 0 or 64
    const int wn   = (wave & 3) * 32;        // 0,32,64,96
    const int tile_t = blockIdx.x * BM;      // token tile fastest-varying
    const int tile_n = blockIdx.y * BN;

    // staging registers for global->LDS
    float4 xr[4];
    v8h    gr[2], ur[2];

    auto load_global = [&](int k0) {
#pragma unroll
        for (int i = 0; i < 4; ++i) {
            int idx = tid + i * 256;             // 0..1023
            int r = idx >> 3;                    // 128 rows, 8 float4/row
            int c = (idx & 7) * 4;
            xr[i] = *(const float4*)(x + (long long)(tile_t + r) * K + k0 + c);
        }
#pragma unroll
        for (int i = 0; i < 2; ++i) {
            int idx = tid + i * 256;             // 0..511
            int r = idx >> 2;                    // 128 rows, 4 v8h/row
            int c = (idx & 3) * 8;
            gr[i] = *(const v8h*)(Wg + (long long)(tile_n + r) * K + k0 + c);
            ur[i] = *(const v8h*)(Wu + (long long)(tile_n + r) * K + k0 + c);
        }
    };
    auto store_lds = [&](int b) {
#pragma unroll
        for (int i = 0; i < 4; ++i) {
            int idx = tid + i * 256;
            int r = idx >> 3;
            int c = (idx & 7) * 4;
            v4h hv = { (_Float16)xr[i].x, (_Float16)xr[i].y,
                       (_Float16)xr[i].z, (_Float16)xr[i].w };
            *(v4h*)(&xs[b][r * LDK + c]) = hv;
        }
#pragma unroll
        for (int i = 0; i < 2; ++i) {
            int idx = tid + i * 256;
            int r = idx >> 2;
            int c = (idx & 3) * 8;
            *(v8h*)(&gs[b][r * LDK + c]) = gr[i];
            *(v8h*)(&us[b][r * LDK + c]) = ur[i];
        }
    };

    v8f zero = {0.f, 0.f, 0.f, 0.f, 0.f, 0.f, 0.f, 0.f};
    v8f accg[4][2], accu[4][2];
#pragma unroll
    for (int i = 0; i < 4; ++i)
#pragma unroll
        for (int j = 0; j < 2; ++j) { accg[i][j] = zero; accu[i][j] = zero; }

    const int NK = K / BK;
    load_global(0);
    store_lds(0);
    __syncthreads();

    int buf = 0;
    for (int kt = 0; kt < NK; ++kt) {
        if (kt + 1 < NK) load_global((kt + 1) * BK);

        v16h a[4], bg[2], bu[2];
#pragma unroll
        for (int i = 0; i < 4; ++i) a[i]  = load_frag(xs[buf], wm + i * 16, lane);
#pragma unroll
        for (int j = 0; j < 2; ++j) {
            bg[j] = load_frag(gs[buf], wn + j * 16, lane);
            bu[j] = load_frag(us[buf], wn + j * 16, lane);
        }
#pragma unroll
        for (int i = 0; i < 4; ++i)
#pragma unroll
            for (int j = 0; j < 2; ++j) {
                accg[i][j] = wmma16(a[i], bg[j], accg[i][j]);
                accu[i][j] = wmma16(a[i], bu[j], accu[i][j]);
            }

        if (kt + 1 < NK) {
            store_lds(buf ^ 1);
            __syncthreads();
            buf ^= 1;
        }
    }

    // Epilogue: silu(g)*u with hardware rcp. D layout: M=i+8*(lane/16), N=lane%16
    const int rbase = (lane >> 4) * 8;
    const int col   = lane & 15;
#pragma unroll
    for (int mi = 0; mi < 4; ++mi)
#pragma unroll
        for (int ni = 0; ni < 2; ++ni)
#pragma unroll
            for (int i = 0; i < 8; ++i) {
                float g = accg[mi][ni][i];
                float u = accu[mi][ni][i];
                float sig = __builtin_amdgcn_rcpf(1.0f + __expf(-g));
                float hv = g * sig * u;           // silu(g)*u
                int t = tile_t + wm + mi * 16 + rbase + i;
                int n = tile_n + wn + ni * 16 + col;
                h[(long long)t * M + n] = (_Float16)hv;
            }
}

// ---------------------------------------------------------------------------
// Down GEMM: A f16 [T,K], W f16 [N,K] -> out f32 [T,N]
// Same tiling/pipeline as above, single weight matrix.
// ---------------------------------------------------------------------------
__global__ __launch_bounds__(256) void gemm_down_kernel(
    const _Float16* __restrict__ A, const _Float16* __restrict__ W,
    float* __restrict__ out, int T, int K, int N)
{
    __shared__ _Float16 as[2][BM * LDK];
    __shared__ _Float16 ws[2][BN * LDK];

    const int tid  = threadIdx.x;
    const int lane = tid & 31;
    const int wave = tid >> 5;
    const int wm   = (wave >> 2) * 64;
    const int wn   = (wave & 3) * 32;
    const int tile_t = blockIdx.x * BM;
    const int tile_n = blockIdx.y * BN;

    v8h ar[2], wr[2];
    auto load_global = [&](int k0) {
#pragma unroll
        for (int i = 0; i < 2; ++i) {
            int idx = tid + i * 256;
            int r = idx >> 2;
            int c = (idx & 3) * 8;
            ar[i] = *(const v8h*)(A + (long long)(tile_t + r) * K + k0 + c);
            wr[i] = *(const v8h*)(W + (long long)(tile_n + r) * K + k0 + c);
        }
    };
    auto store_lds = [&](int b) {
#pragma unroll
        for (int i = 0; i < 2; ++i) {
            int idx = tid + i * 256;
            int r = idx >> 2;
            int c = (idx & 3) * 8;
            *(v8h*)(&as[b][r * LDK + c]) = ar[i];
            *(v8h*)(&ws[b][r * LDK + c]) = wr[i];
        }
    };

    v8f zero = {0.f, 0.f, 0.f, 0.f, 0.f, 0.f, 0.f, 0.f};
    v8f acc[4][2];
#pragma unroll
    for (int i = 0; i < 4; ++i)
#pragma unroll
        for (int j = 0; j < 2; ++j) acc[i][j] = zero;

    const int NK = K / BK;
    load_global(0);
    store_lds(0);
    __syncthreads();

    int buf = 0;
    for (int kt = 0; kt < NK; ++kt) {
        if (kt + 1 < NK) load_global((kt + 1) * BK);

        v16h a[4], b[2];
#pragma unroll
        for (int i = 0; i < 4; ++i) a[i] = load_frag(as[buf], wm + i * 16, lane);
#pragma unroll
        for (int j = 0; j < 2; ++j) b[j] = load_frag(ws[buf], wn + j * 16, lane);
#pragma unroll
        for (int i = 0; i < 4; ++i)
#pragma unroll
            for (int j = 0; j < 2; ++j)
                acc[i][j] = wmma16(a[i], b[j], acc[i][j]);

        if (kt + 1 < NK) {
            store_lds(buf ^ 1);
            __syncthreads();
            buf ^= 1;
        }
    }

    const int rbase = (lane >> 4) * 8;
    const int col   = lane & 15;
#pragma unroll
    for (int mi = 0; mi < 4; ++mi)
#pragma unroll
        for (int ni = 0; ni < 2; ++ni)
#pragma unroll
            for (int i = 0; i < 8; ++i) {
                int t = tile_t + wm + mi * 16 + rbase + i;
                int n = tile_n + wn + ni * 16 + col;
                out[(long long)t * N + n] = acc[mi][ni][i];
            }
}

// ---------------------------------------------------------------------------
extern "C" void kernel_launch(void* const* d_in, const int* in_sizes, int n_in,
                              void* d_out, int out_size, void* d_ws, size_t ws_size,
                              hipStream_t stream)
{
    (void)in_sizes; (void)n_in; (void)out_size; (void)ws_size;

    const float* x           = (const float*)d_in[0];
    const int*   gate_codes  = (const int*)d_in[1];
    const float* gate_scales = (const float*)d_in[2];
    const int*   up_codes    = (const int*)d_in[3];
    const float* up_scales   = (const float*)d_in[4];
    const int*   down_codes  = (const int*)d_in[5];
    const float* down_scales = (const float*)d_in[6];
    float* out = (float*)d_out;

    // Workspace layout (f16): Wg [M,HD], Wu [M,HD], Wd [HD,M], h [T,M]
    char* ws = (char*)d_ws;
    size_t wbytes = (size_t)MDIM * HD * sizeof(_Float16);   // 90.2 MB each
    _Float16* Wg = (_Float16*)(ws);
    _Float16* Wu = (_Float16*)(ws + wbytes);
    _Float16* Wd = (_Float16*)(ws + 2 * wbytes);
    _Float16* h  = (_Float16*)(ws + 3 * wbytes);            // 180 MB

    // 1) Dequant all three weights to f16
    int nElem = MDIM * HD;                     // 45,088,768
    int dqBlocks = nElem / (256 * 4);
    dequant_nf4_kernel<<<dqBlocks, 256, 0, stream>>>(
        gate_codes, gate_scales, Wg, HD, HD / QBLK);
    dequant_nf4_kernel<<<dqBlocks, 256, 0, stream>>>(
        up_codes, up_scales, Wu, HD, HD / QBLK);
    dequant_nf4_kernel<<<dqBlocks, 256, 0, stream>>>(
        down_codes, down_scales, Wd, MDIM, MDIM / QBLK);

    // 2) Fused gate/up GEMM + SwiGLU -> h   (token tile = blockIdx.x for L2 reuse)
    dim3 g1(T_TOK / BM, MDIM / BN);            // 64 x 86
    gemm_gateup_kernel<<<g1, 256, 0, stream>>>(x, Wg, Wu, h, T_TOK, HD, MDIM);

    // 3) Down projection -> out
    dim3 g2(T_TOK / BM, HD / BN);              // 64 x 32
    gemm_down_kernel<<<g2, 256, 0, stream>>>(h, Wd, out, T_TOK, MDIM, HD);
}
